// DenseCapsule_3891240370330
// MI455X (gfx1250) — compile-verified, matrix operand order checked by reading, not run on hardware
//
#include <hip/hip_runtime.h>

// Problem constants (from reference setup_inputs)
#define B_    32
#define NI_   2048
#define DI_   16
#define NO_   64
#define DO_   32
#define NCOL  (NO_ * DO_)     // 2048 columns of u_hat per (b,i)
#define KTOT  (NI_ * DI_)     // 32768 : contraction length of the round-0 GEMM
#define NIC   8               // capsules per workgroup in k_route

typedef __attribute__((ext_vector_type(16))) __bf16 v16bf;
typedef __attribute__((ext_vector_type(8)))  float  v8f;
typedef unsigned int       u32;
typedef unsigned long long u64;
typedef __attribute__((ext_vector_type(4))) u32 u32x4;
typedef __attribute__((ext_vector_type(8))) u32 u32x8;

// fp32 -> bf16 with round-to-nearest-even
__device__ __forceinline__ __bf16 f32_bf16(float f) {
    union { float f; unsigned u; } in; in.f = f;
    unsigned u = in.u + 0x7FFFu + ((in.u >> 16) & 1u);
    union { unsigned short s; __bf16 h; } out; out.s = (unsigned short)(u >> 16);
    return out.h;
}

// ISA 16-bit A-matrix 16x32 layout: VGPR g holds K = base, base+1 per lane.
// lanes 0-15: base = 2g (g<4) else 16+2(g-4);  lanes 16-31: +8
__device__ __forceinline__ int kbase_of(int g, int hi) {
    return ((g < 4) ? (2 * g) : (16 + 2 * (g - 4))) + (hi ? 8 : 0);
}

// ---------------------------------------------------------------------------
// Tensor Data Mover: contiguous 1-D copy of nElem fp32 words global -> LDS.
// D# group0: count=1, lds_addr, global_addr[56:0], type=2.
// D# group1: data_size=4B, tensor_dim0=tile_dim0=nElem, dim1 unused.
// Issued once per wave (caller guards to wave 0); tracked by TENSORcnt.
__device__ __forceinline__ void tdm_load_1d(u64 gaddr, u32 lds_byte_off, u32 nElem) {
    u32x4 g0;
    g0[0] = 1u;                                            // count=1, user mode
    g0[1] = lds_byte_off;                                  // lds_addr (bytes)
    g0[2] = (u32)(gaddr & 0xFFFFFFFFu);                    // global_addr[31:0]
    g0[3] = ((u32)(gaddr >> 32) & 0x01FFFFFFu) | (2u << 30); // [56:32] | type=2
    u32x8 g1;
    g1[0] = 2u << 16;                                      // data_size=2 (4 bytes)
    g1[1] = (nElem & 0xFFFFu) << 16;                       // tensor_dim0[15:0]
    g1[2] = (nElem >> 16) | (1u << 16);                    // dim0[31:16], dim1=1
    g1[3] = (nElem & 0xFFFFu) << 16;                       // tile_dim0
    g1[4] = 0u;                                            // tile_dim1/2 unused
    g1[5] = nElem;                                         // tensor_dim0_stride
    g1[6] = 0u;
    g1[7] = 0u;
    asm volatile("tensor_load_to_lds %0, %1" :: "s"(g0), "s"(g1) : "memory");
}

// ---------------------------------------------------------------------------
__global__ void k_zero(float* __restrict__ p, int n) {
    int idx = blockIdx.x * blockDim.x + threadIdx.x;
    if (idx < n) p[idx] = 0.f;
}

// ---------------------------------------------------------------------------
// Round-0 s:  s[b, j*32+e] = sum_{i,d} x[b,i,d] * W[i,j,d,e]
// One big GEMM  X(32 x 32768) @ Wflat(32768 x 2048), split-K with fp32 atomics.
__global__ void k_uhat_s0(const float* __restrict__ x,
                          const float* __restrict__ W,
                          float* __restrict__ s) {
    const int tid  = threadIdx.x;
    const int lane = tid & 31;
    const int w    = blockIdx.x * 8 + (tid >> 5);   // global wave id, 0..8191
    const int ntile = w & 127;
    const int ks    = w >> 7;                        // 0..63
    const int n  = lane & 15;
    const int hi = lane >> 4;
    const int col = ntile * 16 + n;                  // output column (j*32+e)
    const int j = col >> 5, e = col & 31;
    const long wcol = (long)j * (DI_ * DO_) + e;     // j*512 + e

    v8f acc0 = {}, acc1 = {};
    for (int step = 0; step < 16; ++step) {
        const int kb = ks * 512 + step * 32;
        v16bf a0, a1, bm;
#pragma unroll
        for (int g = 0; g < 8; ++g) {
            const int kk = kb + kbase_of(g, hi);     // even => kk+1 same capsule
            const float* xp0 = x + (long)n        * KTOT + kk;
            const float* xp1 = x + (long)(16 + n) * KTOT + kk;
            a0[2*g]   = f32_bf16(xp0[0]);  a0[2*g+1] = f32_bf16(xp0[1]);
            a1[2*g]   = f32_bf16(xp1[0]);  a1[2*g+1] = f32_bf16(xp1[1]);
            const int icap = kk >> 4, d = kk & 15;
            const float* wp = W + (long)icap * 32768 + wcol + (long)d * 32;
            __builtin_prefetch(wp + 64, 0, 1);
            bm[2*g]   = f32_bf16(wp[0]);
            bm[2*g+1] = f32_bf16(wp[32]);
        }
        acc0 = __builtin_amdgcn_wmma_f32_16x16x32_bf16(false, a0, false, bm,
                                                       (short)0, acc0, false, false);
        acc1 = __builtin_amdgcn_wmma_f32_16x16x32_bf16(false, a1, false, bm,
                                                       (short)0, acc1, false, false);
    }
#pragma unroll
    for (int r = 0; r < 8; ++r) {
        const int m = r + hi * 8;
        atomicAdd(&s[(long)m        * NCOL + col], acc0[r]);
        atomicAdd(&s[(long)(16 + m) * NCOL + col], acc1[r]);
    }
}

// ---------------------------------------------------------------------------
// v = squash(s * scale); re-zeroes s for the next accumulation pass.
__global__ void k_squash(float* __restrict__ s, float* __restrict__ v, float scale) {
    const int tid  = threadIdx.x;
    const int lane = tid & 31;
    const int row  = blockIdx.x * 8 + (tid >> 5);    // 0..2047 = b*64+j
    const int idx  = row * DO_ + lane;
    float val = s[idx] * scale;
    float sq  = val * val;
#pragma unroll
    for (int off = 16; off > 0; off >>= 1) sq += __shfl_xor(sq, off, 32);
    v[idx] = val * (sq / (1.f + sq)) * rsqrtf(sq + 1e-9f);
    s[idx] = 0.f;
}

// ---------------------------------------------------------------------------
// Fused routing pass, round r.  grid=256, blockDim=1024 (32 waves), 8 capsules
// per workgroup.  For each capsule: TDM-stream W[i] into double-buffered LDS
// (8 x 16KB chunks), recompute u_hat (32x2048) into LDS via bf16 WMMA,
// agreement + b-logit update, softmax over j, accumulate c*u_hat into a
// per-thread 64-float register accumulator; one atomic flush per workgroup.
// LDS: wstg 32KB | uh 256KB | xs 2KB | bnew 8KB | cmat 8KB = 306KB (< 320KB).
__global__ void k_route(const float* __restrict__ x, const float* __restrict__ W,
                        const float* __restrict__ v, float* __restrict__ blog,
                        float* __restrict__ s, int round) {
    extern __shared__ float sm[];
    float* wstg = sm;                        // [2][4096]  (byte off 0 / 16384)
    float* uh   = sm + 8192;                 // [32][2048]
    float* xs   = uh + B_ * NCOL;            // [32][16]
    float* bnew = xs + B_ * DI_;             // [32][64]
    float* cmat = bnew + B_ * NO_;           // [32][64]

    const int tid  = threadIdx.x;
    const int lane = tid & 31;
    const int wv   = tid >> 5;               // 0..31
    const int n    = lane & 15;
    const int hi   = lane >> 4;
    const int mt   = wv & 1;                 // M-tile of this wave
    const int nt   = wv >> 1;                // N-tile within a 256-col chunk

    float sacc[64];
#pragma unroll
    for (int q = 0; q < 64; ++q) sacc[q] = 0.f;

    for (int cap = 0; cap < NIC; ++cap) {
        const int i = blockIdx.x * NIC + cap;
        const u64 wbase = (u64)(const void*)(W + (long)i * 32768);

        // --- phase 1: stage x[:, i, :] (32x16); wave0 kicks off chunk 0 DMA ---
        if (wv == 0 && lane == 0) tdm_load_1d(wbase, 0u, 4096u);
        for (int t = tid; t < B_ * DI_; t += 1024) {
            const int b = t >> 4, d = t & 15;
            xs[t] = x[((long)b * NI_ + i) * DI_ + d];
        }
        __syncthreads();

        // A fragment for this wave's M-tile (K padded 16->32 with zeros)
        v16bf a;
#pragma unroll
        for (int g = 0; g < 8; ++g) {
            const int kb = kbase_of(g, hi);
            if (kb < 16) {
                a[2*g]   = f32_bf16(xs[(mt * 16 + n) * 16 + kb]);
                a[2*g+1] = f32_bf16(xs[(mt * 16 + n) * 16 + kb + 1]);
            } else { a[2*g] = a[2*g+1] = (__bf16)0.f; }
        }

        // --- phase 2: 8 pipelined chunks: TDM W chunk -> LDS, WMMA -> uh ---
        for (int c = 0; c < 8; ++c) {
            if (wv == 0) {
                if (lane == 0 && c < 7)
                    tdm_load_1d(wbase + (u64)(c + 1) * 16384u,
                                (u32)(((c + 1) & 1) * 16384u), 4096u);
                if (c < 7) __builtin_amdgcn_s_wait_tensorcnt(1);
                else       __builtin_amdgcn_s_wait_tensorcnt(0);
            }
            __syncthreads();                           // chunk c visible to all

            const float* wbuf = wstg + (c & 1) * 4096; // [jj 0..7][d 0..15][e 0..31]
            const int lc = nt * 16 + n;                // local col in chunk
            const int jj = lc >> 5, e = lc & 31;
            v16bf bm;
#pragma unroll
            for (int g = 0; g < 8; ++g) {
                const int kb = kbase_of(g, hi);
                if (kb < 16) {
                    bm[2*g]   = f32_bf16(wbuf[jj * 512 + kb * 32 + e]);
                    bm[2*g+1] = f32_bf16(wbuf[jj * 512 + (kb + 1) * 32 + e]);
                } else { bm[2*g] = bm[2*g+1] = (__bf16)0.f; }
            }
            v8f acc = {};
            acc = __builtin_amdgcn_wmma_f32_16x16x32_bf16(false, a, false, bm,
                                                          (short)0, acc, false, false);
            const int col = c * 256 + nt * 16 + n;
#pragma unroll
            for (int r = 0; r < 8; ++r)
                uh[(mt * 16 + r + hi * 8) * NCOL + col] = acc[r];
            __syncthreads();                           // done reading wstg[c&1]
        }

        // --- phase 3: agreement + b-logit update ---
        for (int p = tid; p < B_ * NO_; p += 1024) {
            const int b = p >> 6, j = p & 63;
            const float* uhr = uh + b * NCOL + j * DO_;
            const float* vr  = v + ((long)b * NO_ + j) * DO_;
            float agree = 0.f;
#pragma unroll
            for (int e = 0; e < DO_; ++e) agree += uhr[e] * vr[e];
            const long bi = ((long)b * NI_ + i) * NO_ + j;
            const float bn = (round == 0 ? 0.f : blog[bi]) + agree;
            blog[bi] = bn;
            bnew[p]  = bn;
        }
        __syncthreads();

        // --- phase 4: softmax over j per batch row ---
        if (tid < B_) {
            const int b = tid;
            float mx = -1e30f;
            for (int j = 0; j < NO_; ++j) mx = fmaxf(mx, bnew[b * NO_ + j]);
            float sum = 0.f;
            for (int j = 0; j < NO_; ++j) sum += __expf(bnew[b * NO_ + j] - mx);
            const float inv = 1.f / sum;
            for (int j = 0; j < NO_; ++j)
                cmat[b * NO_ + j] = __expf(bnew[b * NO_ + j] - mx) * inv;
        }
        __syncthreads();

        // --- phase 5: accumulate c * u_hat into registers (no atomics) ---
#pragma unroll
        for (int q = 0; q < 64; ++q) {
            const int p = tid + 1024 * q;
            const int b = p >> 11, rem = p & 2047, j = rem >> 5;
            sacc[q] += cmat[b * NO_ + j] * uh[b * NCOL + rem];
        }
        __syncthreads();                               // before next cap reuses LDS
    }

    // --- single atomic flush: 64 coalesced fp32 atomics per thread ---
#pragma unroll
    for (int q = 0; q < 64; ++q)
        atomicAdd(&s[tid + 1024 * q], sacc[q]);
}

// ---------------------------------------------------------------------------
extern "C" void kernel_launch(void* const* d_in, const int* in_sizes, int n_in,
                              void* d_out, int out_size, void* d_ws, size_t ws_size,
                              hipStream_t stream) {
    const float* x = (const float*)d_in[0];   // (32, 2048, 16)
    const float* W = (const float*)d_in[1];   // (2048, 64, 16, 32)
    float* out = (float*)d_out;               // (32, 64, 32) = 65536 floats

    const int SNE = B_ * NO_ * DO_;           // 65536
    const long BNE = (long)B_ * NI_ * NO_;    // 4,194,304
    const size_t need = ((size_t)SNE * 2 + (size_t)BNE) * sizeof(float);
    if (ws_size < need) return;

    float* s_buf = (float*)d_ws;              // 256 KB
    float* v_buf = s_buf + SNE;               // 256 KB
    float* b_buf = v_buf + SNE;               // 16 MB

    const size_t route_lds =
        (size_t)(8192 + B_ * NCOL + B_ * DI_ + 2 * B_ * NO_) * sizeof(float); // 306KB

    k_zero<<<SNE / 256, 256, 0, stream>>>(s_buf, SNE);
    // round 0: s0 = sum_i u_hat (uniform coupling folded into squash scale)
    k_uhat_s0<<<1024, 256, 0, stream>>>(x, W, s_buf);
    k_squash<<<256, 256, 0, stream>>>(s_buf, v_buf, 1.f / 64.f);   // v0, s<-0
    // round 0 update + round 1 s-accumulation (fused, recomputes u_hat)
    k_route<<<NI_ / NIC, 1024, route_lds, stream>>>(x, W, v_buf, b_buf, s_buf, 0);
    k_squash<<<256, 256, 0, stream>>>(s_buf, v_buf, 1.f);          // v1, s<-0
    // round 1 update + round 2 s-accumulation
    k_route<<<NI_ / NIC, 1024, route_lds, stream>>>(x, W, v_buf, b_buf, s_buf, 1);
    k_squash<<<256, 256, 0, stream>>>(s_buf, out, 1.f);            // v2 -> d_out
}
